// SABGATEBlock_89154931130444
// MI455X (gfx1250) — matrix operand order repaired and checked by reading, not compiled, not run on hardware
//
#include <hip/hip_runtime.h>
#include <hip/hip_bf16.h>
#include <math.h>

typedef __bf16 bf16;
typedef __attribute__((ext_vector_type(2)))  __bf16 bf16x2;
typedef __attribute__((ext_vector_type(4)))  __bf16 bf16x4;
typedef __attribute__((ext_vector_type(8)))  __bf16 bf16x8;
typedef __attribute__((ext_vector_type(16))) __bf16 v16bf;
typedef __attribute__((ext_vector_type(8)))  float  v8f;

#define WMMA_BF16(A, Bv, Cv) \
  __builtin_amdgcn_wmma_f32_16x16x32_bf16(false, (A), false, (Bv), (short)0, (Cv), false, false)

// ---------------------------------------------------------------------------
// WMMA fragment loaders (CDNA5 wave32 16-bit layouts, CDNA5 ISA).
// Each lane's 16-element fragment is two contiguous, 16B-aligned bf16x8 chunks
// -> two ds_load_b128 per fragment.
// A tile: row-major As[m][k], lda = 40 (80B rows). B tile: TRANSPOSED
// BsT[n][k], ldb = 40. C/D: VGPR r -> m = r + 8h, n = lane&15.
// ---------------------------------------------------------------------------
__device__ __forceinline__ v16bf load_a_frag(const bf16* As, int lda, int lane) {
  const int m = lane & 15;
  const int h = (lane >> 4) & 1;
  const bf16* row = As + m * lda;
  bf16x8 lo = *(const bf16x8*)(row + h * 8);
  bf16x8 hi = *(const bf16x8*)(row + 16 + h * 8);
  return __builtin_shufflevector(lo, hi, 0, 1, 2, 3, 4, 5, 6, 7,
                                 8, 9, 10, 11, 12, 13, 14, 15);
}

__device__ __forceinline__ v16bf load_b_frag(const bf16* BsT, int ldb, int lane) {
  const int n = lane & 15;
  const int h = (lane >> 4) & 1;
  const bf16* row = BsT + n * ldb;
  bf16x8 lo = *(const bf16x8*)(row + h * 8);
  bf16x8 hi = *(const bf16x8*)(row + 16 + h * 8);
  return __builtin_shufflevector(lo, hi, 0, 1, 2, 3, 4, 5, 6, 7,
                                 8, 9, 10, 11, 12, 13, 14, 15);
}

__device__ __forceinline__ bf16x4 cvt4(float4 f) {
  bf16x4 v;
  v[0] = (bf16)f.x; v[1] = (bf16)f.y; v[2] = (bf16)f.z; v[3] = (bf16)f.w;
  return v;
}

// packed convert + single 4B-aligned ds_store_b32 (k even)
__device__ __forceinline__ void st_pk(bf16* dst, float a, float b) {
  bf16x2 v;
  v[0] = (bf16)a; v[1] = (bf16)b;
  *(bf16x2*)dst = v;
}

__device__ __forceinline__ float gelu_exact(float v) {
  return 0.5f * v * (1.0f + erff(v * 0.70710678118654752f));
}

// ---------------------------------------------------------------------------
// B-tile loader: 32k x 64n slab of X -> BsT[n][k].  Two-phase (all global
// loads issued first, then packed converts + b32 stores of k-pairs).
// thread i-th pair idx: kp = idx>>4 (k = 2*kp), n4 = idx&15 (n = 4*n4).
// ---------------------------------------------------------------------------
__device__ __forceinline__ void load_btile_32x64(bf16 (*BsT)[40], const float* __restrict__ Xp,
                                                 int k0, int nb, int N, int tid) {
  float4 f0[2], f1[2];
#pragma unroll
  for (int i = 0; i < 2; ++i) {
    int idx = tid + 128 * i;
    int kp = idx >> 4, n4 = idx & 15;
    const float* p = Xp + (size_t)(k0 + 2 * kp) * N + nb + n4 * 4;
    f0[i] = *(const float4*)(p);
    f1[i] = *(const float4*)(p + N);
  }
#pragma unroll
  for (int i = 0; i < 2; ++i) {
    int idx = tid + 128 * i;
    int kp = idx >> 4, n4 = idx & 15;
    int n = n4 * 4, k = kp * 2;
    st_pk(&BsT[n + 0][k], f0[i].x, f1[i].x);
    st_pk(&BsT[n + 1][k], f0[i].y, f1[i].y);
    st_pk(&BsT[n + 2][k], f0[i].z, f1[i].z);
    st_pk(&BsT[n + 3][k], f0[i].w, f1[i].w);
  }
}

// ---------------------------------------------------------------------------
// Generic channel-mixing GEMM: Out[b](Cout x N) = act(W x X[b] + bias) + resid
// W: (Cout x Cin) row-major, or (Cin x Cout) if WTRANS. 64x64 tile / block,
// 128 threads (4 waves), bf16 WMMA with f32 accumulate.
// grid = (N/64, Cout/64, B)
// ---------------------------------------------------------------------------
template <int WTRANS>
__global__ __launch_bounds__(128)
void gemm_cn_kernel(const float* __restrict__ Wm, const float* __restrict__ X,
                    const float* __restrict__ bias, const float* __restrict__ resid,
                    float* __restrict__ Out,
                    int Cout, int Cin, int N, int wBatchStride, int act)
{
  __shared__ bf16 As[64][40];    // [m][k]
  __shared__ bf16 BsT[64][40];   // [n][k]  (transposed tile)
  const int tid  = threadIdx.x;
  const int lane = tid & 31;
  const int wv   = tid >> 5;
  const int nb   = blockIdx.x * 64;
  const int mb   = blockIdx.y * 64;
  const int bb   = blockIdx.z;
  const float* Wp = Wm + (size_t)bb * wBatchStride;
  const float* Xp = X + (size_t)bb * Cin * N;

  v8f z = {};
  v8f acc[4];
#pragma unroll
  for (int t = 0; t < 4; ++t) acc[t] = z;

  for (int k0 = 0; k0 < Cin; k0 += 32) {
    if (!WTRANS) {
      float4 a[4];
#pragma unroll
      for (int i = 0; i < 4; ++i) {         // A tile 64x32: 512 float4, batched
        int idx = tid + 128 * i;
        int r = idx >> 3, c4 = idx & 7;
        a[i] = *(const float4*)(Wp + (size_t)(mb + r) * Cin + k0 + c4 * 4);
      }
#pragma unroll
      for (int i = 0; i < 4; ++i) {
        int idx = tid + 128 * i;
        int r = idx >> 3, c4 = idx & 7;
        *(bf16x4*)&As[r][c4 * 4] = cvt4(a[i]);
      }
    } else {
      float4 g0[2], g1[2];                  // read along m (contiguous), k-pairs
#pragma unroll
      for (int i = 0; i < 2; ++i) {
        int idx = tid + 128 * i;
        int kp = idx >> 4, m4 = idx & 15;
        const float* p = Wp + (size_t)(k0 + 2 * kp) * Cout + mb + m4 * 4;
        g0[i] = *(const float4*)(p);
        g1[i] = *(const float4*)(p + Cout);
      }
#pragma unroll
      for (int i = 0; i < 2; ++i) {
        int idx = tid + 128 * i;
        int kp = idx >> 4, m4 = idx & 15;
        int m = m4 * 4, k = kp * 2;
        st_pk(&As[m + 0][k], g0[i].x, g1[i].x);
        st_pk(&As[m + 1][k], g0[i].y, g1[i].y);
        st_pk(&As[m + 2][k], g0[i].z, g1[i].z);
        st_pk(&As[m + 3][k], g0[i].w, g1[i].w);
      }
    }
    load_btile_32x64(BsT, Xp, k0, nb, N, tid);
    __syncthreads();
    v16bf a = load_a_frag(&As[wv * 16][0], 40, lane);
#pragma unroll
    for (int t = 0; t < 4; ++t) {
      v16bf b = load_b_frag(&BsT[t * 16][0], 40, lane);
      acc[t] = WMMA_BF16(a, b, acc[t]);
    }
    __syncthreads();
  }

  const int h = lane >> 4, nn = lane & 15;
#pragma unroll
  for (int t = 0; t < 4; ++t) {
#pragma unroll
    for (int r = 0; r < 8; ++r) {
      int m = mb + wv * 16 + r + 8 * h;
      int n = nb + t * 16 + nn;
      float v = acc[t][r];
      if (bias)  v += bias[m];
      if (act == 1) v = gelu_exact(v);
      size_t o = ((size_t)bb * Cout + m) * N + n;
      if (resid) v += resid[o];
      Out[o] = v;
    }
  }
}

// ---------------------------------------------------------------------------
// K1: logits = coef_w(64x256) x xf + coef_b; Alpha = softmax over k.
// grid = (N/64, B), 128 threads.
// ---------------------------------------------------------------------------
__global__ __launch_bounds__(128)
void coef_softmax_kernel(const float* __restrict__ x, const float* __restrict__ coef_w,
                         const float* __restrict__ coef_b, float* __restrict__ Alpha,
                         int C, int N)
{
  __shared__ bf16  As[64][40];
  __shared__ bf16  BsT[64][40];
  __shared__ float Ls[64][65];
  const int tid  = threadIdx.x;
  const int lane = tid & 31;
  const int wv   = tid >> 5;
  const int nb   = blockIdx.x * 64;
  const int bb   = blockIdx.y;
  const float* Xp = x + (size_t)bb * C * N;

  v8f z = {};
  v8f acc[4];
#pragma unroll
  for (int t = 0; t < 4; ++t) acc[t] = z;

  for (int k0 = 0; k0 < C; k0 += 32) {
    float4 a[4];
#pragma unroll
    for (int i = 0; i < 4; ++i) {
      int idx = tid + 128 * i;
      int r = idx >> 3, c4 = idx & 7;
      a[i] = *(const float4*)(coef_w + (size_t)r * C + k0 + c4 * 4);
    }
#pragma unroll
    for (int i = 0; i < 4; ++i) {
      int idx = tid + 128 * i;
      int r = idx >> 3, c4 = idx & 7;
      *(bf16x4*)&As[r][c4 * 4] = cvt4(a[i]);
    }
    load_btile_32x64(BsT, Xp, k0, nb, N, tid);
    __syncthreads();
    v16bf a2 = load_a_frag(&As[wv * 16][0], 40, lane);
#pragma unroll
    for (int t = 0; t < 4; ++t) {
      v16bf b = load_b_frag(&BsT[t * 16][0], 40, lane);
      acc[t] = WMMA_BF16(a2, b, acc[t]);
    }
    __syncthreads();
  }

  const int h = lane >> 4, nn = lane & 15;
#pragma unroll
  for (int t = 0; t < 4; ++t)
#pragma unroll
    for (int r = 0; r < 8; ++r)
      Ls[wv * 16 + r + 8 * h][t * 16 + nn] = acc[t][r];
  __syncthreads();

  if (tid < 64) {
    const int col = tid;
    float mx = -3.0e38f;
    for (int k = 0; k < 64; ++k) mx = fmaxf(mx, Ls[k][col] + coef_b[k]);
    float sum = 0.f;
    for (int k = 0; k < 64; ++k) {
      float e = expf(Ls[k][col] + coef_b[k] - mx);
      Ls[k][col] = e;
      sum += e;
    }
    float inv = 1.f / sum;
    for (int k = 0; k < 64; ++k)
      Alpha[((size_t)bb * 64 + k) * N + nb + col] = Ls[k][col] * inv;
  }
}

// ---------------------------------------------------------------------------
// K3: Vf = pool_w(16x256) x xrec + pool_b;  L = U(128x16) x Vf;  S = softmax_m(L)
// grid = (N/64, B), 128 threads. Vf tile via WMMA, L + softmax scalar in LDS.
// ---------------------------------------------------------------------------
__global__ __launch_bounds__(128)
void pool_softmax_kernel(const float* __restrict__ xrec, const float* __restrict__ pool_w,
                         const float* __restrict__ pool_b, const float* __restrict__ U,
                         float* __restrict__ S, int C, int N)
{
  __shared__ bf16  As[16][40];
  __shared__ bf16  BsT[64][40];
  __shared__ float Vs[16][66];
  __shared__ float Ls[128][65];
  const int tid  = threadIdx.x;
  const int lane = tid & 31;
  const int wv   = tid >> 5;
  const int nb   = blockIdx.x * 64;
  const int bb   = blockIdx.y;
  const float* Xp = xrec + (size_t)bb * C * N;

  v8f acc = {};
  for (int k0 = 0; k0 < C; k0 += 32) {
    {                                       // A tile 16x32: 128 float4
      int r = tid >> 3, c4 = tid & 7;
      float4 f = *(const float4*)(pool_w + (size_t)r * C + k0 + c4 * 4);
      *(bf16x4*)&As[r][c4 * 4] = cvt4(f);
    }
    load_btile_32x64(BsT, Xp, k0, nb, N, tid);
    __syncthreads();
    v16bf a = load_a_frag(&As[0][0], 40, lane);
    v16bf b = load_b_frag(&BsT[wv * 16][0], 40, lane);
    acc = WMMA_BF16(a, b, acc);
    __syncthreads();
  }

  const int h = lane >> 4, nn = lane & 15;
#pragma unroll
  for (int r = 0; r < 8; ++r)
    Vs[r + 8 * h][wv * 16 + nn] = acc[r] + pool_b[r + 8 * h];
  __syncthreads();

  { // L[m][col] = sum_r U[m][r] * Vf[r][col], one row m per thread
    const int m = tid;
    float u[16];
#pragma unroll
    for (int r = 0; r < 16; ++r) u[r] = U[m * 16 + r];
    for (int col = 0; col < 64; ++col) {
      float s = 0.f;
#pragma unroll
      for (int r = 0; r < 16; ++r) s += u[r] * Vs[r][col];
      Ls[m][col] = s;
    }
  }
  __syncthreads();

  if (tid < 64) {
    const int col = tid;
    float mx = -3.0e38f;
    for (int m = 0; m < 128; ++m) mx = fmaxf(mx, Ls[m][col]);
    float sum = 0.f;
    for (int m = 0; m < 128; ++m) { float e = expf(Ls[m][col] - mx); Ls[m][col] = e; sum += e; }
    float inv = 1.f / sum;
    for (int m = 0; m < 128; ++m)
      S[((size_t)bb * 128 + m) * N + nb + col] = Ls[m][col] * inv;
  }
}

// ---------------------------------------------------------------------------
// K4: T[b](C1 x C2) = X1[b](C1 x N) x X2[b](C2 x N)^T  (K-dim = N)
// grid = (C1/64, B), 128 threads, wave strip 16 x C2(=128) = 8 WMMA tiles.
// Transposed-LDS B layout: both global read and LDS write fully contiguous.
// ---------------------------------------------------------------------------
__global__ __launch_bounds__(128)
void gemm_abt_kernel(const float* __restrict__ X1, const float* __restrict__ X2,
                     float* __restrict__ Out, int C1, int C2, int N)
{
  __shared__ bf16 As[64][40];     // [c1][n-k]
  __shared__ bf16 BsT[128][40];   // [c2][n-k]
  const int tid  = threadIdx.x;
  const int lane = tid & 31;
  const int wv   = tid >> 5;
  const int mb   = blockIdx.x * 64;
  const int bb   = blockIdx.y;
  const float* x1 = X1 + (size_t)bb * C1 * N;
  const float* x2 = X2 + (size_t)bb * C2 * N;

  v8f z = {};
  v8f acc[8];
#pragma unroll
  for (int t = 0; t < 8; ++t) acc[t] = z;

  for (int n0 = 0; n0 < N; n0 += 32) {
    float4 a[4], q[8];
#pragma unroll
    for (int i = 0; i < 4; ++i) {           // A tile 64x32 from X1 (batched)
      int idx = tid + 128 * i;
      int r = idx >> 3, c4 = idx & 7;
      a[i] = *(const float4*)(x1 + (size_t)(mb + r) * N + n0 + c4 * 4);
    }
    {                                       // B tile: one contiguous 32f row / thread
      const float* src = x2 + (size_t)tid * N + n0;
#pragma unroll
      for (int j = 0; j < 8; ++j) q[j] = *(const float4*)(src + j * 4);
    }
#pragma unroll
    for (int i = 0; i < 4; ++i) {
      int idx = tid + 128 * i;
      int r = idx >> 3, c4 = idx & 7;
      *(bf16x4*)&As[r][c4 * 4] = cvt4(a[i]);
    }
#pragma unroll
    for (int j = 0; j < 4; ++j) {
      bf16x8 v;
      v[0] = (bf16)q[2*j].x; v[1] = (bf16)q[2*j].y;
      v[2] = (bf16)q[2*j].z; v[3] = (bf16)q[2*j].w;
      v[4] = (bf16)q[2*j+1].x; v[5] = (bf16)q[2*j+1].y;
      v[6] = (bf16)q[2*j+1].z; v[7] = (bf16)q[2*j+1].w;
      *(bf16x8*)&BsT[tid][j * 8] = v;
    }
    __syncthreads();
    v16bf af = load_a_frag(&As[wv * 16][0], 40, lane);
#pragma unroll
    for (int t = 0; t < 8; ++t) {
      v16bf b = load_b_frag(&BsT[t * 16][0], 40, lane);
      acc[t] = WMMA_BF16(af, b, acc[t]);
    }
    __syncthreads();
  }

  const int h = lane >> 4, nn = lane & 15;
#pragma unroll
  for (int t = 0; t < 8; ++t)
#pragma unroll
    for (int r = 0; r < 8; ++r) {
      int m = mb + wv * 16 + r + 8 * h;
      int c2 = t * 16 + nn;
      Out[((size_t)bb * C1 + m) * C2 + c2] = acc[t][r];
    }
}

// ---------------------------------------------------------------------------
// K5a: small attention core, one block per batch, 256 threads, scalar math.
// Produces That[b](C x M).
// ---------------------------------------------------------------------------
__global__ __launch_bounds__(256)
void attn_kernel(const float* __restrict__ T, const float* __restrict__ Wq,
                 const float* __restrict__ Wk, const float* __restrict__ Wv,
                 const float* __restrict__ Rfull, const float* __restrict__ Wout,
                 float* __restrict__ That, int C, int Mm, int D)
{
  __shared__ float sm[14336];
  float* sQ  = sm;           // 8192 floats, reused as O
  float* sTR = sm + 8192;    // 4096 floats, reused as softmax A (2048)
  float* sKp = sm + 12288;   // 1024
  float* sVp = sm + 13312;   // 1024
  const int bb = blockIdx.x, tid = threadIdx.x;
  const float* Tb = T + (size_t)bb * C * Mm;

  for (int i = tid; i < Mm * D; i += 256) {           // Q = Tm @ Wq
    int m = i / D, d = i % D;
    float s = 0.f;
    for (int c = 0; c < C; ++c) s += Tb[c * Mm + m] * Wq[c * D + d];
    sQ[i] = s;
  }
  for (int i = tid; i < 16 * C; i += 256) {           // TR[r][c] = sum_m Rm[r][m] T[c][m]
    int r = i / C, c = i % C;
    float s = 0.f;
    for (int m = 0; m < Mm; ++m) s += Rfull[r * Mm + m] * Tb[c * Mm + m];
    sTR[i] = s;
  }
  __syncthreads();
  for (int i = tid; i < 16 * D; i += 256) {           // Kp, Vp = TR @ Wk/Wv
    int r = i / D, d = i % D;
    float sk = 0.f, sv = 0.f;
    for (int c = 0; c < C; ++c) {
      float t = sTR[r * C + c];
      sk += t * Wk[c * D + d];
      sv += t * Wv[c * D + d];
    }
    sKp[i] = sk; sVp[i] = sv;
  }
  __syncthreads();
  float* sA = sTR;                                    // scores softmax over r
  if (tid < Mm) {
    float sc[16];
    float mx = -3.0e38f;
#pragma unroll
    for (int r = 0; r < 16; ++r) {
      float s = 0.f;
      for (int d = 0; d < D; ++d) s += sQ[tid * D + d] * sKp[r * D + d];
      s *= 0.125f;                                    // 1/sqrt(64)
      sc[r] = s; mx = fmaxf(mx, s);
    }
    float sum = 0.f;
#pragma unroll
    for (int r = 0; r < 16; ++r) { sc[r] = expf(sc[r] - mx); sum += sc[r]; }
    float inv = 1.f / sum;
#pragma unroll
    for (int r = 0; r < 16; ++r) sA[tid * 16 + r] = sc[r] * inv;
  }
  __syncthreads();
  float* sO = sQ;                                     // O = A @ Vp (overwrite Q)
  for (int i = tid; i < Mm * D; i += 256) {
    int m = i / D, d = i % D;
    float s = 0.f;
#pragma unroll
    for (int r = 0; r < 16; ++r) s += sA[m * 16 + r] * sVp[r * D + d];
    sO[i] = s;
  }
  __syncthreads();
  for (int i = tid; i < C * Mm; i += 256) {           // That[c][m] = sum_d O[m][d] Wout[d][c]
    int c = i / Mm, m = i % Mm;
    float s = 0.f;
    for (int d = 0; d < D; ++d) s += sO[m * D + d] * Wout[d * C + c];
    That[(size_t)bb * C * Mm + i] = s;
  }
}

// ---------------------------------------------------------------------------
// K5b: Ut = Pu^T-project(That); W2 = B_mat @ Ut. One block per batch.
// ---------------------------------------------------------------------------
__global__ __launch_bounds__(256)
void proj_kernel(const float* __restrict__ That, const float* __restrict__ Pu,
                 const float* __restrict__ Bmat, float* __restrict__ W2,
                 int C, int Mm, int Kk)
{
  __shared__ float sUt[64 * 128];
  const int bb = blockIdx.x, tid = threadIdx.x;
  const float* Tb = That + (size_t)bb * C * Mm;
  for (int i = tid; i < Kk * Mm; i += 256) {          // Ut[k][m] = sum_c That[c][m] Pu[c][k]
    int k = i / Mm, m = i % Mm;
    float s = 0.f;
    for (int c = 0; c < C; ++c) s += Tb[c * Mm + m] * Pu[c * Kk + k];
    sUt[i] = s;
  }
  __syncthreads();
  for (int i = tid; i < C * Mm; i += 256) {           // W2[c][m] = sum_k Bmat[c][k] Ut[k][m]
    int c = i / Mm, m = i % Mm;
    float s = 0.f;
    for (int k = 0; k < Kk; ++k) s += Bmat[c * Kk + k] * sUt[k * Mm + m];
    W2[(size_t)bb * C * Mm + i] = s;
  }
}

// ---------------------------------------------------------------------------
// K6c: gating.  g = sigmoid(gate_w . XA + gate_b); Xout = g*XA + (1-g)*XB
// one thread per (b, n)
// ---------------------------------------------------------------------------
__global__ __launch_bounds__(256)
void gate_kernel(const float* __restrict__ XA, const float* __restrict__ XB,
                 const float* __restrict__ gate_w, const float* __restrict__ gate_b,
                 float* __restrict__ Xout, int C, int N)
{
  int idx = blockIdx.x * blockDim.x + threadIdx.x;   // b*N + n
  int b = idx / N, n = idx - b * N;
  const float* xa = XA + (size_t)b * C * N + n;
  const float* xb = XB + (size_t)b * C * N + n;
  float s = 0.f;
  for (int c = 0; c < C; ++c) s += gate_w[c] * xa[(size_t)c * N];
  float g = 1.f / (1.f + expf(-(s + gate_b[0])));
  float* xo = Xout + (size_t)b * C * N + n;
  for (int c = 0; c < C; ++c)
    xo[(size_t)c * N] = g * xa[(size_t)c * N] + (1.f - g) * xb[(size_t)c * N];
}

// ---------------------------------------------------------------------------
// K7a: depthwise 3x3 SAME conv, one thread per output element
// ---------------------------------------------------------------------------
__global__ __launch_bounds__(256)
void dwconv_kernel(const float* __restrict__ X, const float* __restrict__ w,
                   const float* __restrict__ bch, float* __restrict__ Y,
                   int C, int H, int W)
{
  int idx = blockIdx.x * blockDim.x + threadIdx.x;
  int wq = idx % W; int t = idx / W;
  int hq = t % H;  t /= H;
  int c  = t % C;  int b = t / C;
  const float* xp = X + (((size_t)b * C + c) * H) * W;
  const float* wp = w + c * 9;
  float s = 0.f;
#pragma unroll
  for (int ky = 0; ky < 3; ++ky) {
    int hh = hq + ky - 1;
    if (hh < 0 || hh >= H) continue;
#pragma unroll
    for (int kx = 0; kx < 3; ++kx) {
      int ww = wq + kx - 1;
      if (ww < 0 || ww >= W) continue;
      s += wp[ky * 3 + kx] * xp[(size_t)hh * W + ww];
    }
  }
  Y[idx] = s + bch[c];
}

// ---------------------------------------------------------------------------
// K8a: LayerNorm over channel dim (per b,n)
// ---------------------------------------------------------------------------
__global__ __launch_bounds__(256)
void ln_kernel(const float* __restrict__ X, const float* __restrict__ g,
               const float* __restrict__ be, float* __restrict__ Y, int C, int N)
{
  int idx = blockIdx.x * blockDim.x + threadIdx.x;   // b*N + n
  int b = idx / N, n = idx - b * N;
  const float* xp = X + (size_t)b * C * N + n;
  float s = 0.f, s2 = 0.f;
  for (int c = 0; c < C; ++c) { float v = xp[(size_t)c * N]; s += v; s2 += v * v; }
  float mu = s / (float)C;
  float var = s2 / (float)C - mu * mu;
  float r = rsqrtf(var + 1e-5f);
  float* yp = Y + (size_t)b * C * N + n;
  for (int c = 0; c < C; ++c)
    yp[(size_t)c * N] = (xp[(size_t)c * N] - mu) * r * g[c] + be[c];
}

// ---------------------------------------------------------------------------
// Host orchestration
// ---------------------------------------------------------------------------
extern "C" void kernel_launch(void* const* d_in, const int* in_sizes, int n_in,
                              void* d_out, int out_size, void* d_ws, size_t ws_size,
                              hipStream_t stream) {
  (void)in_sizes; (void)n_in; (void)out_size; (void)ws_size;
  const float* x      = (const float*)d_in[0];
  const float* coef_w = (const float*)d_in[1];
  const float* coef_b = (const float*)d_in[2];
  const float* B_mat  = (const float*)d_in[3];
  const float* pool_w = (const float*)d_in[4];
  const float* pool_b = (const float*)d_in[5];
  const float* U      = (const float*)d_in[6];
  const float* Wq     = (const float*)d_in[7];
  const float* Wk     = (const float*)d_in[8];
  const float* Wv     = (const float*)d_in[9];
  const float* Wout   = (const float*)d_in[10];
  const float* R_full = (const float*)d_in[11];
  const float* Pu     = (const float*)d_in[12];
  const float* gate_w = (const float*)d_in[13];
  const float* gate_b = (const float*)d_in[14];
  const float* dw_w   = (const float*)d_in[15];
  const float* dw_b   = (const float*)d_in[16];
  const float* pw_w   = (const float*)d_in[17];
  const float* pw_b   = (const float*)d_in[18];
  const float* ln_g   = (const float*)d_in[19];
  const float* ln_b   = (const float*)d_in[20];
  const float* fc1_w  = (const float*)d_in[21];
  const float* fc1_b  = (const float*)d_in[22];
  const float* fc2_w  = (const float*)d_in[23];
  const float* fc2_b  = (const float*)d_in[24];
  const float* fuse_w = (const float*)d_in[25];
  const float* fuse_b = (const float*)d_in[26];

  const int Bn = 16, C = 256, N = 4096, Mm = 128, Kk = 64, HID = 512;
  const size_t SZ_H1 = (size_t)Bn * HID * N;   // 33,554,432 floats
  const size_t SZ_S  = (size_t)Bn * Mm * N;    //  8,388,608
  const size_t SZ_CN = (size_t)Bn * C * N;     // 16,777,216
  const size_t SZ_T  = (size_t)Bn * C * Mm;    //     524,288

  float* ws    = (float*)d_ws;
  float* H1    = ws;                           // also hosts ALPHA + XREC (dead before fc1)
  float* ALPHA = ws;                           // B x 64 x N
  float* XREC  = ws + (size_t)Bn * Kk * N;     // B x 256 x N, inside H1 region
  float* SBUF  = ws + SZ_H1;                   // S: B x 128 x N
  float* XOUT  = SBUF + SZ_S;                  // Xout
  float* BUFA  = XOUT + SZ_CN;                 // XA -> dw -> XLN -> OutSum
  float* BUFB  = BUFA + SZ_CN;                 // XB -> pw
  float* TBUF  = BUFB + SZ_CN;                 // T
  float* THAT  = TBUF + SZ_T;                  // That
  float* W2B   = THAT + SZ_T;                  // B_mat @ Ut

  // 1) Alpha = softmax_k(coef_w x + coef_b)
  coef_softmax_kernel<<<dim3(N / 64, Bn), 128, 0, stream>>>(x, coef_w, coef_b, ALPHA, C, N);
  // 2) xrec = B_mat @ Alpha
  gemm_cn_kernel<0><<<dim3(N / 64, C / 64, Bn), 128, 0, stream>>>(
      B_mat, ALPHA, nullptr, nullptr, XREC, C, Kk, N, 0, 0);
  // 3) S = softmax_m(U @ (pool_w @ xrec + pool_b))
  pool_softmax_kernel<<<dim3(N / 64, Bn), 128, 0, stream>>>(XREC, pool_w, pool_b, U, SBUF, C, N);
  // 4) T = xrec @ S^T
  gemm_abt_kernel<<<dim3(C / 64, Bn), 128, 0, stream>>>(XREC, SBUF, TBUF, C, Mm, N);
  // 5) attention core -> That ; 6) W2 = B_mat @ Ut
  attn_kernel<<<Bn, 256, 0, stream>>>(TBUF, Wq, Wk, Wv, R_full, Wout, THAT, C, Mm, 64);
  proj_kernel<<<Bn, 256, 0, stream>>>(THAT, Pu, B_mat, W2B, C, Mm, Kk);
  // 7) XA = That @ S ; XB = W2 @ S   (per-batch W)
  gemm_cn_kernel<0><<<dim3(N / 64, C / 64, Bn), 128, 0, stream>>>(
      THAT, SBUF, nullptr, nullptr, BUFA, C, Mm, N, C * Mm, 0);
  gemm_cn_kernel<0><<<dim3(N / 64, C / 64, Bn), 128, 0, stream>>>(
      W2B, SBUF, nullptr, nullptr, BUFB, C, Mm, N, C * Mm, 0);
  // 8) gating -> Xout
  gate_kernel<<<(Bn * N) / 256, 256, 0, stream>>>(BUFA, BUFB, gate_w, gate_b, XOUT, C, N);
  // 9) depthwise conv -> dw (BUFA)
  dwconv_kernel<<<(Bn * C * 64 * 64) / 256, 256, 0, stream>>>(XOUT, dw_w, dw_b, BUFA, C, 64, 64);
  // 10) pw = pw_w @ dw + pw_b -> BUFB
  gemm_cn_kernel<0><<<dim3(N / 64, C / 64, Bn), 128, 0, stream>>>(
      pw_w, BUFA, pw_b, nullptr, BUFB, C, C, N, 0, 0);
  // 11) XLN = LayerNorm_c(Xout) -> BUFA
  ln_kernel<<<(Bn * N) / 256, 256, 0, stream>>>(XOUT, ln_g, ln_b, BUFA, C, N);
  // 12) H1 = gelu(fc1_w^T @ XLN + fc1_b)
  gemm_cn_kernel<1><<<dim3(N / 64, HID / 64, Bn), 128, 0, stream>>>(
      fc1_w, BUFA, fc1_b, nullptr, H1, HID, C, N, 0, 1);
  // 13) OutSum = fc2_w^T @ H1 + fc2_b + pw -> BUFA
  gemm_cn_kernel<1><<<dim3(N / 64, C / 64, Bn), 128, 0, stream>>>(
      fc2_w, H1, fc2_b, BUFB, BUFA, C, HID, N, 0, 0);
  // 14) out = fuse_w @ OutSum + fuse_b + Xout -> d_out
  gemm_cn_kernel<0><<<dim3(N / 64, C / 64, Bn), 128, 0, stream>>>(
      fuse_w, BUFA, fuse_b, XOUT, (float*)d_out, C, C, N, 0, 0);
}
